// Wasserstain_70231305224565
// MI455X (gfx1250) — matrix operand order repaired and checked by reading, not compiled, not run on hardware
//
#include <hip/hip_runtime.h>
#include <hip/hip_bf16.h>

typedef __attribute__((ext_vector_type(8)))  __bf16 v8bf;
typedef __attribute__((ext_vector_type(16))) __bf16 v16bf;
typedef __attribute__((ext_vector_type(8)))  float  v8f;

#define DIMD   256
#define RV     36     // valid image regions
#define WV     50     // valid caption words
#define RPAD   48     // 3 x 16 m-tiles
#define WPAD   64     // 4 x 16 n-tiles
#define PITCH  65     // LDS row pitch (floats) -> conflict-free rows & cols
#define NB     128    // batch (both sides)
#define LAMB_INV 20.0f
#define EPSV   1e-6f
#define WAVES  4      // pairs per block (1 wave32 per pair)

__device__ __forceinline__ float wave_sum(float v) {
  #pragma unroll
  for (int m = 16; m >= 1; m >>= 1) v += __shfl_xor(v, m, 32);
  return v;
}

// ---------------------------------------------------------------------------
// Kernel 1: L2-normalize each row, convert to bf16, write padded (pad rows=0).
// One block per item, 4 waves, one wave per row (round-robin).
// ---------------------------------------------------------------------------
__global__ __launch_bounds__(128)
void normalize_pad(const float* __restrict__ src, unsigned short* __restrict__ dstRaw,
                   int rowsValid, int rowsPad) {
  __bf16* dst = reinterpret_cast<__bf16*>(dstRaw);
  const int lane = threadIdx.x & 31;
  const int wave = threadIdx.x >> 5;
  const int item = blockIdx.x;
  const float* s = src + (size_t)item * rowsValid * DIMD;
  __bf16*      d = dst + (size_t)item * rowsPad   * DIMD;

  for (int r = wave; r < rowsPad; r += 4) {
    v8bf o;
    if (r < rowsValid) {
      float x[8]; float ss = 0.f;
      #pragma unroll
      for (int t = 0; t < 8; ++t) { x[t] = s[r * DIMD + lane * 8 + t]; ss += x[t] * x[t]; }
      ss = wave_sum(ss);
      const float inv = 1.0f / fmaxf(sqrtf(ss), 1e-8f);
      #pragma unroll
      for (int t = 0; t < 8; ++t) o[t] = (__bf16)(x[t] * inv);
    } else {
      #pragma unroll
      for (int t = 0; t < 8; ++t) o[t] = (__bf16)0.0f;
    }
    *reinterpret_cast<v8bf*>(d + r * DIMD + lane * 8) = o;
  }
}

// ---------------------------------------------------------------------------
// Kernel 2: one wave per (i,j) pair.
//   S[48x64] = imgsN[i] (48x256) x capsN[j]^T  via v_wmma_f32_16x16x32_bf16
//   Sinkhorn (3 iters) on masked 36x50 tile in LDS, sims = sum(S*P).
// ---------------------------------------------------------------------------
__global__ __launch_bounds__(128)
void sinkhorn_pairs(const unsigned short* __restrict__ imgsRaw,
                    const unsigned short* __restrict__ capsRaw,
                    const int* __restrict__ img_lens,
                    const int* __restrict__ cap_lens,
                    float* __restrict__ out) {
  __shared__ float lds[WAVES][PITCH * RV + RV + WPAD];  // P tile + U[36] + V[64]

  const __bf16* imgsP = reinterpret_cast<const __bf16*>(imgsRaw);
  const __bf16* capsP = reinterpret_cast<const __bf16*>(capsRaw);

  const int lane = threadIdx.x & 31;
  const int wave = threadIdx.x >> 5;
  const int pair = blockIdx.x * WAVES + wave;
  const int i = pair >> 7;        // /128
  const int j = pair & (NB - 1);  // %128

  float* P = &lds[wave][0];
  float* U = P + PITCH * RV;
  float* V = U + RV;

  const __bf16* Abase = imgsP + (size_t)i * RPAD * DIMD;
  const __bf16* Bbase = capsP + (size_t)j * WPAD * DIMD;

  const int hi = lane >> 4;   // half-wave id
  const int lo = lane & 15;

  v8f acc[3][4];
  #pragma unroll
  for (int mt = 0; mt < 3; ++mt)
    #pragma unroll
    for (int nt = 0; nt < 4; ++nt)
      acc[mt][nt] = (v8f){0.f, 0.f, 0.f, 0.f, 0.f, 0.f, 0.f, 0.f};

  // ---- GEMM: K = 256 in steps of 32 ----
  #pragma unroll
  for (int ks = 0; ks < DIMD; ks += 32) {
    v16bf a[3], b[4];
    #pragma unroll
    for (int mt = 0; mt < 3; ++mt) {
      // A 16-bit 16x32 frag: lane<16 -> K ks..ks+15; lane>=16 -> K ks+16..ks+31
      const __bf16* pa = Abase + (size_t)(mt * 16 + lo) * DIMD + ks + hi * 16;
      a[mt] = *reinterpret_cast<const v16bf*>(pa);
    }
    #pragma unroll
    for (int nt = 0; nt < 4; ++nt) {
      // B 16-bit 32x16 frag: N = lo, lanes 0-15 hold K=ks..ks+15, lanes 16-31 K=ks+16..ks+31
      const __bf16* pb = Bbase + (size_t)(nt * 16 + lo) * DIMD + ks + hi * 16;
      b[nt] = *reinterpret_cast<const v16bf*>(pb);
    }
    #pragma unroll
    for (int mt = 0; mt < 3; ++mt)
      #pragma unroll
      for (int nt = 0; nt < 4; ++nt)
        acc[mt][nt] = __builtin_amdgcn_wmma_f32_16x16x32_bf16(
            false, a[mt], false, b[nt], (short)0, acc[mt][nt], false, false);
  }

  // ---- Sinkhorn on the 36x50 valid region ----
  const int Ri = img_lens[i];
  const int Wj = cap_lens[j];
  const float invR = 1.0f / (float)Ri;
  const float invW = 1.0f / (float)Wj;

  // init P = mask ? exp((S-1)/lamb) : 0 ; track global sum
  float psum = 0.f;
  #pragma unroll
  for (int mt = 0; mt < 3; ++mt)
    #pragma unroll
    for (int nt = 0; nt < 4; ++nt)
      #pragma unroll
      for (int k = 0; k < 8; ++k) {
        const int m = mt * 16 + k + hi * 8;   // C/D layout: VGPR k, halves offset by 8
        const int n = nt * 16 + lo;
        if (m < RV) {
          const float s = acc[mt][nt][k];
          const float p = (m < Ri && n < Wj) ? __expf((s - 1.0f) * LAMB_INV) : 0.0f;
          P[m * PITCH + n] = p;
          psum += p;
        }
      }
  const float pinv = 1.0f / (wave_sum(psum) + EPSV);
  __syncthreads();

  // normalize P globally
  for (int idx = lane; idx < RV * WPAD; idx += 32) {
    const int m = idx >> 6, n = idx & 63;
    P[m * PITCH + n] *= pinv;
  }
  __syncthreads();

  #pragma unroll 1
  for (int it = 0; it < 3; ++it) {
    // row sums -> row factors
    for (int m = lane; m < RV; m += 32) {
      float s = 0.f;
      for (int n = 0; n < WV; ++n) s += P[m * PITCH + n];
      U[m] = ((m < Ri) ? invR : 0.0f) / (s + EPSV);
    }
    __syncthreads();
    for (int idx = lane; idx < RV * WPAD; idx += 32) {
      const int m = idx >> 6, n = idx & 63;
      P[m * PITCH + n] *= U[m];
    }
    __syncthreads();
    // col sums -> col factors
    for (int n = lane; n < WPAD; n += 32) {
      if (n < WV) {
        float s = 0.f;
        for (int m = 0; m < RV; ++m) s += P[m * PITCH + n];
        V[n] = ((n < Wj) ? invW : 0.0f) / (s + EPSV);
      } else {
        V[n] = 0.0f;
      }
    }
    __syncthreads();
    for (int idx = lane; idx < RV * WPAD; idx += 32) {
      const int m = idx >> 6, n = idx & 63;
      P[m * PITCH + n] *= V[n];
    }
    __syncthreads();
  }

  // sims = sum(S * P) ; masked entries have P==0
  float fsum = 0.f;
  #pragma unroll
  for (int mt = 0; mt < 3; ++mt)
    #pragma unroll
    for (int nt = 0; nt < 4; ++nt)
      #pragma unroll
      for (int k = 0; k < 8; ++k) {
        const int m = mt * 16 + k + hi * 8;
        const int n = nt * 16 + lo;
        if (m < RV) fsum += acc[mt][nt][k] * P[m * PITCH + n];
      }
  fsum = wave_sum(fsum);
  if (lane == 0) out[i * NB + j] = fsum;
}

extern "C" void kernel_launch(void* const* d_in, const int* in_sizes, int n_in,
                              void* d_out, int out_size, void* d_ws, size_t ws_size,
                              hipStream_t stream) {
  const float* imgs     = (const float*)d_in[0];
  const float* caps     = (const float*)d_in[1];
  const int*   img_lens = (const int*)d_in[2];
  const int*   cap_lens = (const int*)d_in[3];
  float*       out      = (float*)d_out;

  unsigned short* imgsP = (unsigned short*)d_ws;
  unsigned short* capsP = imgsP + (size_t)NB * RPAD * DIMD;  // bf16 elements

  normalize_pad<<<NB, 128, 0, stream>>>(imgs, imgsP, RV, RPAD);
  normalize_pad<<<NB, 128, 0, stream>>>(caps, capsP, WV, WPAD);
  sinkhorn_pairs<<<(NB * NB) / WAVES, 32 * WAVES, 0, stream>>>(
      imgsP, capsP, img_lens, cap_lens, out);
}
